// GShardMoE_27736898797645
// MI455X (gfx1250) — matrix operand order repaired and checked by reading, not compiled, not run on hardware
//
#include <hip/hip_runtime.h>
#include <hip/hip_bf16.h>

// ---------------------------------------------------------------------------
// GShard top-2 MoE for MI455X / gfx1250.
//   pre-pass : x -> bf16; all weights -> bf16 *transposed* [N][K] (once/call)
//   router   : wave-per-token logits -> softmax -> top2 -> token lists
//   FFN GEMMs: 128x128x32 tiles, v_wmma_f32_16x16x32_bf16, tiles staged by
//              the Tensor Data Mover (tensor_load_to_lds) with a
//              double-buffered pipeline (DMA of tile it+1 overlaps WMMA of
//              tile it; s_wait_tensorcnt at the top of each iteration).
//              Gathered A tiles (routed experts) staged with b128 copies.
// ---------------------------------------------------------------------------

typedef __bf16 bf16_t;
typedef __attribute__((ext_vector_type(8)))  bf16_t v8bf;
typedef __attribute__((ext_vector_type(16))) bf16_t v16bf;
typedef __attribute__((ext_vector_type(8)))  float  v8f;

typedef unsigned int uint32x4 __attribute__((ext_vector_type(4)));
typedef int          int32x4  __attribute__((ext_vector_type(4)));
typedef int          int32x8  __attribute__((ext_vector_type(8)));

#define BM 128
#define BN 128
#define BK 32
#define ABUF (BM * BK)            // halves per A tile (8 KB)
#define BUFH (BM * BK + BN * BK)  // halves per double-buffer slot (16 KB)

// ---- TDM availability / arity detection -----------------------------------
#if defined(__has_builtin)
#if __has_builtin(__builtin_amdgcn_tensor_load_to_lds)
#define HAVE_TDM 1
#endif
#endif
#if __has_include(<hip/amd_detail/amd_gfx1250_TDM.h>) || \
    (defined(__clang_major__) && __clang_major__ >= 23)
#define TDM_ARGS6 1
#endif

__device__ __forceinline__ unsigned short f2bf(float f) {
  unsigned int u = __float_as_uint(f);
  u += 0x7FFFu + ((u >> 16) & 1u);          // round-to-nearest-even
  return (unsigned short)(u >> 16);
}

__device__ __forceinline__ float gelu_tanh(float v) {
  const float c0 = 0.7978845608028654f;     // sqrt(2/pi)
  float v3 = v * v * v;
  return 0.5f * v * (1.0f + tanhf(c0 * (v + 0.044715f * v3)));
}

#if defined(HAVE_TDM)
// 2-D tile DMA: global (bf16, row stride `strideElems`) -> LDS, tile
// tileK x tileRows, rows contiguous in LDS (tileK*2 bytes apart).
__device__ __forceinline__ void tdm_load_2d(unsigned ldsOff, const void* gptr,
                                            unsigned tileK, unsigned tileRows,
                                            unsigned long long strideElems) {
  unsigned long long ga = (unsigned long long)(size_t)gptr;
  uint32x4 g0;
  g0[0] = 1u;                                        // count=1 (valid D#)
  g0[1] = ldsOff;                                    // lds_addr
  g0[2] = (unsigned)(ga & 0xffffffffu);              // global_addr lo
  g0[3] = (unsigned)((ga >> 32) & 0x01ffffffu) | 0x80000000u;  // hi | type=2
  const unsigned td0 = 0x7fffffffu, td1 = 0x7fffffffu;  // huge dims: no OOB
  int32x8 g1;
  g1[0] = (int)0x00010000u;                          // data_size=1 (2 bytes)
  g1[1] = (int)((td0 & 0xffffu) << 16);              // tensor_dim0 lo16
  g1[2] = (int)((td0 >> 16) | ((td1 & 0xffffu) << 16));
  g1[3] = (int)((td1 >> 16) | (tileK << 16));        // tile_dim0
  g1[4] = (int)(tileRows & 0xffffu);                 // tile_dim1, tile_dim2=0
  g1[5] = (int)(unsigned)(strideElems & 0xffffffffu);// dim0_stride lo32
  g1[6] = (int)(unsigned)((strideElems >> 32) & 0xffffu);
  g1[7] = 0;
  int32x4 z4 = {0, 0, 0, 0};
#if defined(TDM_ARGS6)
  int32x8 z8 = {0, 0, 0, 0, 0, 0, 0, 0};
  __builtin_amdgcn_tensor_load_to_lds(g0, g1, z4, z4, z8, 0);
#else
  __builtin_amdgcn_tensor_load_to_lds(g0, g1, z4, z4, 0);
#endif
}
#endif

// manual staging fallback: 128x32-half tile, b128 copies
__device__ __forceinline__ void stage_tile_128x32(unsigned short* lds,
                                                  const unsigned short* g,
                                                  int rowBase, int k0, int K,
                                                  int t) {
  int r = t >> 1;
  int c = (t & 1) * 16;
  const int4* p = (const int4*)&g[(size_t)(rowBase + r) * K + k0 + c];
  *(int4*)&lds[r * BK + c]     = p[0];
  *(int4*)&lds[r * BK + c + 8] = p[1];
}

// ----------------------------- pre-passes ----------------------------------
__global__ __launch_bounds__(256)
void cvt_x_k(const float* __restrict__ x, unsigned short* __restrict__ xb) {
  size_t i = (size_t)blockIdx.x * 256 + threadIdx.x;
  const float4* src = (const float4*)x;
  float4 v = src[i];
  ushort4 o;
  o.x = f2bf(v.x); o.y = f2bf(v.y); o.z = f2bf(v.z); o.w = f2bf(v.w);
  *(ushort4*)&xb[i * 4] = o;
}

// W [K][N] f32  ->  WT [N][K] bf16 ; z==0: shared matrix, z>=1: expert z-1
__global__ __launch_bounds__(256)
void transpose_cvt_k(const float* __restrict__ shsrc,
                     const float* __restrict__ esrc,
                     unsigned short* __restrict__ shdst,
                     unsigned short* __restrict__ edst, int K, int N) {
  __shared__ unsigned short tl[64 * 72];
  const int z = blockIdx.z;
  const float* src = (z == 0) ? shsrc : esrc + (size_t)(z - 1) * K * N;
  unsigned short* dst = (z == 0) ? shdst : edst + (size_t)(z - 1) * K * N;
  const int k0 = blockIdx.x * 64, n0 = blockIdx.y * 64;
  const int t = threadIdx.x;
#pragma unroll
  for (int j = 0; j < 16; ++j) {
    int e = t + j * 256;
    int kk = e >> 6, nn = e & 63;
    tl[kk * 72 + nn] = f2bf(src[(size_t)(k0 + kk) * N + n0 + nn]);
  }
  __syncthreads();
#pragma unroll
  for (int j = 0; j < 4; ++j) {
    int e = t + j * 256;                   // 1024 chunks of 4 halves
    int nn = e >> 4, kk = (e & 15) * 4;
    ushort4 o;
    o.x = tl[(kk + 0) * 72 + nn];
    o.y = tl[(kk + 1) * 72 + nn];
    o.z = tl[(kk + 2) * 72 + nn];
    o.w = tl[(kk + 3) * 72 + nn];
    *(ushort4*)&dst[(size_t)(n0 + nn) * K + k0 + kk] = o;
  }
}

// ----------------------------- router --------------------------------------
__global__ __launch_bounds__(256)
void moe_router_k(const float* __restrict__ x, const float* __restrict__ gw,
                  const float* __restrict__ gb, float* __restrict__ weights,
                  int* __restrict__ counts, int* __restrict__ tlist, int Ntok,
                  int D) {
  const int wave = threadIdx.x >> 5;
  const int lane = threadIdx.x & 31;
  const int token = blockIdx.x * (blockDim.x >> 5) + wave;
  if (token >= Ntok) return;

  float acc[8];
#pragma unroll
  for (int e = 0; e < 8; ++e) acc[e] = 0.f;
  const float* xr = x + (size_t)token * D;
  for (int d = lane; d < D; d += 32) {
    float xv = xr[d];
    const float* gwr = gw + (size_t)d * 8;
#pragma unroll
    for (int e = 0; e < 8; ++e) acc[e] += xv * gwr[e];
  }
#pragma unroll
  for (int off = 16; off > 0; off >>= 1)
#pragma unroll
    for (int e = 0; e < 8; ++e) acc[e] += __shfl_xor(acc[e], off, 32);

  if (lane == 0) {
    float m = -1e30f;
#pragma unroll
    for (int e = 0; e < 8; ++e) { acc[e] += gb[e]; m = fmaxf(m, acc[e]); }
    float p[8], s = 0.f;
#pragma unroll
    for (int e = 0; e < 8; ++e) { p[e] = __expf(acc[e] - m); s += p[e]; }
    float inv = 1.f / s;
#pragma unroll
    for (int e = 0; e < 8; ++e) p[e] *= inv;
    int i1 = 0;
#pragma unroll
    for (int e = 1; e < 8; ++e) if (p[e] > p[i1]) i1 = e;
    int i2 = (i1 == 0) ? 1 : 0;
#pragma unroll
    for (int e = 0; e < 8; ++e) if (e != i1 && p[e] > p[i2]) i2 = e;
    float denom = p[i1] + p[i2] + 1e-9f;
    weights[(size_t)token * 8 + i1] = p[i1] / denom;
    weights[(size_t)token * 8 + i2] = p[i2] / denom;
    int pos1 = atomicAdd(&counts[i1], 1);
    tlist[(size_t)i1 * Ntok + pos1] = token;
    int pos2 = atomicAdd(&counts[i2], 1);
    tlist[(size_t)i2 * Ntok + pos2] = token;
  }
}

// -------------------- FFN1: H = gelu(Xb @ W1T^T + b1) -----------------------
// Xb bf16 [rows][K] (gathered if GATHER), W1T bf16 [Nn][K], H bf16 [M][Nn]
template <bool GATHER>
__global__ __launch_bounds__(256)
void moe_ffn1_k(const unsigned short* __restrict__ Xb,
                const unsigned short* __restrict__ W1T,
                const float* __restrict__ B1, unsigned short* __restrict__ H,
                const int* __restrict__ gather,
                const int* __restrict__ countPtr, int K, int Nn) {
  __shared__ __align__(128) unsigned short smem[2 * BUFH];  // 32 KB, 2 slots

  int count = 0x7fffffff;
  if (GATHER) {
    count = *countPtr;
    if ((int)blockIdx.x * BM >= count) return;
  }
  const int t = threadIdx.x;
  const int lane = t & 31;
  const int wave = t >> 5;
  const int wm = wave & 1, wn = wave >> 1;
  const int r0 = blockIdx.x * BM;
  const int n0 = blockIdx.y * BN;
  const unsigned smemOff = (unsigned)(size_t)(void*)smem;

  const v8f vz = {0.f, 0.f, 0.f, 0.f, 0.f, 0.f, 0.f, 0.f};
  v8f acc[4][2];
#pragma unroll
  for (int i = 0; i < 4; ++i)
#pragma unroll
    for (int j = 0; j < 2; ++j) acc[i][j] = vz;

  const int c0a = (lane < 16) ? 0 : 16;
  const int kbb = (lane < 16) ? 0 : 16;
  const int mrow = lane & 15;

  // gather row for this thread's staging slice (fixed across K-steps)
  const int myr = t >> 1;
  const int myc = (t & 1) * 16;
  int myrow = -1;
  if (GATHER) {
    int rg = r0 + myr;
    if (rg < count) myrow = gather[rg];
  }

  const int nsteps = K / BK;
#if defined(HAVE_TDM)
  if (wave == 0) {                         // prologue: DMA tile 0 into slot 0
    tdm_load_2d(smemOff + ABUF * 2, W1T + (size_t)n0 * K, BK, BN,
                (unsigned long long)K);
    if (!GATHER)
      tdm_load_2d(smemOff, Xb + (size_t)r0 * K, BK, BM, (unsigned long long)K);
  }
#endif
  for (int it = 0; it < nsteps; ++it) {
    const int k0 = it * BK;
    const int buf = it & 1;
    unsigned short* As = smem + buf * BUFH;
    unsigned short* Bs = As + ABUF;
#if defined(HAVE_TDM)
    if (GATHER) {                          // manual gather of A rows
      int4 v0 = {0, 0, 0, 0}, v1 = {0, 0, 0, 0};
      if (myrow >= 0) {
        const int4* p = (const int4*)&Xb[(size_t)myrow * K + k0 + myc];
        v0 = p[0]; v1 = p[1];
      }
      *(int4*)&As[myr * BK + myc]     = v0;
      *(int4*)&As[myr * BK + myc + 8] = v1;
    }
    if (wave == 0) __builtin_amdgcn_s_wait_tensorcnt(0);
    __syncthreads();
    if (wave == 0 && it + 1 < nsteps) {    // DMA next tile into other slot
      const int k1 = k0 + BK;
      const unsigned nxt = smemOff + (unsigned)(1 - buf) * (BUFH * 2);
      tdm_load_2d(nxt + ABUF * 2, W1T + (size_t)n0 * K + k1, BK, BN,
                  (unsigned long long)K);
      if (!GATHER)
        tdm_load_2d(nxt, Xb + (size_t)r0 * K + k1, BK, BM,
                    (unsigned long long)K);
    }
#else
    stage_tile_128x32(Bs, W1T, n0, k0, K, t);
    if (!GATHER) {
      stage_tile_128x32(As, Xb, r0, k0, K, t);
    } else {
      int4 v0 = {0, 0, 0, 0}, v1 = {0, 0, 0, 0};
      if (myrow >= 0) {
        const int4* p = (const int4*)&Xb[(size_t)myrow * K + k0 + myc];
        v0 = p[0]; v1 = p[1];
      }
      *(int4*)&As[myr * BK + myc]     = v0;
      *(int4*)&As[myr * BK + myc + 8] = v1;
    }
    __syncthreads();
#endif

    v16bf a[4], b[2];
#pragma unroll
    for (int i = 0; i < 4; ++i) {
      int r = wm * 64 + i * 16 + mrow;
      v8bf lo = *reinterpret_cast<const v8bf*>(&As[r * BK + c0a]);
      v8bf hi = *reinterpret_cast<const v8bf*>(&As[r * BK + c0a + 8]);
      a[i] = __builtin_shufflevector(lo, hi, 0, 1, 2, 3, 4, 5, 6, 7, 8, 9, 10,
                                     11, 12, 13, 14, 15);
    }
#pragma unroll
    for (int j = 0; j < 2; ++j) {
      int n = wn * 32 + j * 16 + mrow;
      b[j] = *reinterpret_cast<const v16bf*>(&Bs[n * BK + kbb]);
    }
#pragma unroll
    for (int i = 0; i < 4; ++i)
#pragma unroll
      for (int j = 0; j < 2; ++j)
        acc[i][j] = __builtin_amdgcn_wmma_f32_16x16x32_bf16(
            false, a[i], false, b[j], (short)0, acc[i][j], false, false);
  }

  // epilogue: bias + GELU -> LDS repack -> coalesced b128 stores of H
  const int mhi = (lane >> 4) << 3;
  unsigned short* Cs = smem;               // 64x128 halves = 16 KB
#pragma unroll
  for (int p = 0; p < 2; ++p) {
    __syncthreads();
    if (wm == p) {
#pragma unroll
      for (int i = 0; i < 4; ++i)
#pragma unroll
        for (int j = 0; j < 2; ++j) {
          int colL = wn * 32 + j * 16 + mrow;
          float bias = B1[n0 + colL];
#pragma unroll
          for (int v = 0; v < 8; ++v) {
            int rowL = i * 16 + mhi + v;
            Cs[rowL * BN + colL] = f2bf(gelu_tanh(acc[i][j][v] + bias));
          }
        }
    }
    __syncthreads();
#pragma unroll
    for (int j = 0; j < 4; ++j) {
      int e = t + j * 256;                 // 1024 chunks of 16 B
      int row = e >> 4, cc = (e & 15) * 8;
      int4 vdat = *(const int4*)&Cs[row * BN + cc];
      *(int4*)&H[(size_t)(r0 + p * 64 + row) * Nn + n0 + cc] = vdat;
    }
  }
}

// ---------------- FFN2: Out (+)= (H @ W2T^T + b2) [ + x ] -------------------
// H bf16 [M][K], W2T bf16 [Nn][K]
template <bool GATHER>
__global__ __launch_bounds__(256)
void moe_ffn2_k(const unsigned short* __restrict__ H,
                const unsigned short* __restrict__ W2T,
                const float* __restrict__ B2, const float* __restrict__ Xres,
                float* __restrict__ Out, const int* __restrict__ gather,
                const int* __restrict__ countPtr,
                const float* __restrict__ weights, int expert, int K, int Nn) {
  __shared__ __align__(128) unsigned short smem[2 * BUFH];  // 32 KB

  int count = 0x7fffffff;
  if (GATHER) {
    count = *countPtr;
    if ((int)blockIdx.x * BM >= count) return;
  }
  const int t = threadIdx.x;
  const int lane = t & 31;
  const int wave = t >> 5;
  const int wm = wave & 1, wn = wave >> 1;
  const int r0 = blockIdx.x * BM;
  const int n0 = blockIdx.y * BN;
  const unsigned smemOff = (unsigned)(size_t)(void*)smem;

  const v8f vz = {0.f, 0.f, 0.f, 0.f, 0.f, 0.f, 0.f, 0.f};
  v8f acc[4][2];
#pragma unroll
  for (int i = 0; i < 4; ++i)
#pragma unroll
    for (int j = 0; j < 2; ++j) acc[i][j] = vz;

  const int c0a = (lane < 16) ? 0 : 16;
  const int kbb = (lane < 16) ? 0 : 16;
  const int mrow = lane & 15;

  const int nsteps = K / BK;
#if defined(HAVE_TDM)
  if (wave == 0) {                         // prologue: DMA tile 0 into slot 0
    tdm_load_2d(smemOff, H + (size_t)r0 * K, BK, BM, (unsigned long long)K);
    tdm_load_2d(smemOff + ABUF * 2, W2T + (size_t)n0 * K, BK, BN,
                (unsigned long long)K);
  }
#endif
  for (int it = 0; it < nsteps; ++it) {
    const int k0 = it * BK;
    const int buf = it & 1;
    unsigned short* As = smem + buf * BUFH;
    unsigned short* Bs = As + ABUF;
#if defined(HAVE_TDM)
    if (wave == 0) __builtin_amdgcn_s_wait_tensorcnt(0);
    __syncthreads();
    if (wave == 0 && it + 1 < nsteps) {
      const int k1 = k0 + BK;
      const unsigned nxt = smemOff + (unsigned)(1 - buf) * (BUFH * 2);
      tdm_load_2d(nxt, H + (size_t)r0 * K + k1, BK, BM, (unsigned long long)K);
      tdm_load_2d(nxt + ABUF * 2, W2T + (size_t)n0 * K + k1, BK, BN,
                  (unsigned long long)K);
    }
#else
    stage_tile_128x32(As, H, r0, k0, K, t);
    stage_tile_128x32(Bs, W2T, n0, k0, K, t);
    __syncthreads();
#endif

    v16bf a[4], b[2];
#pragma unroll
    for (int i = 0; i < 4; ++i) {
      int r = wm * 64 + i * 16 + mrow;
      v8bf lo = *reinterpret_cast<const v8bf*>(&As[r * BK + c0a]);
      v8bf hi = *reinterpret_cast<const v8bf*>(&As[r * BK + c0a + 8]);
      a[i] = __builtin_shufflevector(lo, hi, 0, 1, 2, 3, 4, 5, 6, 7, 8, 9, 10,
                                     11, 12, 13, 14, 15);
    }
#pragma unroll
    for (int j = 0; j < 2; ++j) {
      int n = wn * 32 + j * 16 + mrow;
      b[j] = *reinterpret_cast<const v16bf*>(&Bs[n * BK + kbb]);
    }
#pragma unroll
    for (int i = 0; i < 4; ++i)
#pragma unroll
      for (int j = 0; j < 2; ++j)
        acc[i][j] = __builtin_amdgcn_wmma_f32_16x16x32_bf16(
            false, a[i], false, b[j], (short)0, acc[i][j], false, false);
  }

  const int mhi = (lane >> 4) << 3;
  if (!GATHER) {
    // bias + residual via LDS repack -> coalesced b128 stores
    float* Cf = (float*)smem;              // 64x128 f32 = 32 KB
#pragma unroll
    for (int p = 0; p < 2; ++p) {
      __syncthreads();
      if (wm == p) {
#pragma unroll
        for (int i = 0; i < 4; ++i)
#pragma unroll
          for (int j = 0; j < 2; ++j) {
            int colL = wn * 32 + j * 16 + mrow;
            float bias = B2[n0 + colL];
#pragma unroll
            for (int v = 0; v < 8; ++v)
              Cf[(i * 16 + mhi + v) * BN + colL] = acc[i][j][v] + bias;
          }
      }
      __syncthreads();
#pragma unroll
      for (int j = 0; j < 8; ++j) {
        int e = t + j * 256;               // 2048 chunks of 16 B
        int row = e >> 5, cc = (e & 31) * 4;
        size_t g = (size_t)(r0 + p * 64 + row) * Nn + n0 + cc;
        float4 vdat = *(const float4*)&Cf[row * BN + cc];
        float4 xr = *(const float4*)&Xres[g];
        vdat.x += xr.x; vdat.y += xr.y; vdat.z += xr.z; vdat.w += xr.w;
        *(float4*)&Out[g] = vdat;
      }
    }
  } else {
    // weighted scatter-add to routed tokens
#pragma unroll
    for (int i = 0; i < 4; ++i)
#pragma unroll
      for (int j = 0; j < 2; ++j) {
        int col = n0 + wn * 32 + j * 16 + mrow;
        float bias = B2[col];
#pragma unroll
        for (int v = 0; v < 8; ++v) {
          int row = r0 + wm * 64 + i * 16 + mhi + v;
          if (row < count) {
            int tk = gather[row];
            float w = weights[(size_t)tk * 8 + expert];
            atomicAdd(&Out[(size_t)tk * Nn + col], w * (acc[i][j][v] + bias));
          }
        }
      }
  }
}

// ------------------------------ host side ----------------------------------
extern "C" void kernel_launch(void* const* d_in, const int* in_sizes, int n_in,
                              void* d_out, int out_size, void* d_ws,
                              size_t ws_size, hipStream_t stream) {
  (void)in_sizes; (void)n_in; (void)out_size; (void)ws_size;

  const float* x     = (const float*)d_in[0];
  const float* gw    = (const float*)d_in[1];
  const float* gb    = (const float*)d_in[2];
  const float* sh_w1 = (const float*)d_in[3];
  const float* sh_b1 = (const float*)d_in[4];
  const float* sh_w2 = (const float*)d_in[5];
  const float* sh_b2 = (const float*)d_in[6];
  const float* ew1   = (const float*)d_in[7];
  const float* eb1   = (const float*)d_in[8];
  const float* ew2   = (const float*)d_in[9];
  const float* eb2   = (const float*)d_in[10];
  float* out = (float*)d_out;

  const int D = 1024, I = 4096, E = 8;
  const int Ntok = 2 * 2048;
  const size_t DI = (size_t)D * I;

  // workspace carve (~193 MB)
  size_t off = 0;
  auto take = [&](size_t b) {
    size_t p = off;
    off += (b + 255) & ~(size_t)255;
    return p;
  };
  char* ws = (char*)d_ws;
  size_t o_counts  = take(32);
  size_t o_weights = take((size_t)Ntok * 8 * 4);
  size_t o_tlist   = take((size_t)E * Ntok * 4);
  size_t o_xb      = take((size_t)Ntok * D * 2);
  size_t o_h       = take((size_t)Ntok * I * 2);
  size_t o_w1ts    = take(DI * 2);
  size_t o_w2ts    = take(DI * 2);
  size_t o_ew1t    = take((size_t)E * DI * 2);
  size_t o_ew2t    = take((size_t)E * DI * 2);

  int*   counts  = (int*)(ws + o_counts);
  float* weights = (float*)(ws + o_weights);
  int*   tlist   = (int*)(ws + o_tlist);
  unsigned short* Xb   = (unsigned short*)(ws + o_xb);
  unsigned short* Hbuf = (unsigned short*)(ws + o_h);
  unsigned short* W1Ts = (unsigned short*)(ws + o_w1ts);
  unsigned short* W2Ts = (unsigned short*)(ws + o_w2ts);
  unsigned short* EW1T = (unsigned short*)(ws + o_ew1t);
  unsigned short* EW2T = (unsigned short*)(ws + o_ew2t);

  hipMemsetAsync(d_ws, 0, o_tlist, stream);  // zero counts + dense weights

  moe_router_k<<<Ntok / 8, 256, 0, stream>>>(x, gw, gb, weights, counts, tlist,
                                             Ntok, D);
  cvt_x_k<<<(Ntok * D / 4) / 256, 256, 0, stream>>>(x, Xb);
  // W1-type: [D][I] -> [I][D] ; W2-type: [I][D] -> [D][I] ; z=0 shared
  transpose_cvt_k<<<dim3(D / 64, I / 64, E + 1), 256, 0, stream>>>(
      sh_w1, ew1, W1Ts, EW1T, D, I);
  transpose_cvt_k<<<dim3(I / 64, D / 64, E + 1), 256, 0, stream>>>(
      sh_w2, ew2, W2Ts, EW2T, I, D);

  // shared MLP: out = x + gelu(x@w1+b1)@w2+b2
  moe_ffn1_k<false><<<dim3(Ntok / BM, I / BN), 256, 0, stream>>>(
      Xb, W1Ts, sh_b1, Hbuf, nullptr, nullptr, D, I);
  moe_ffn2_k<false><<<dim3(Ntok / BM, D / BN), 256, 0, stream>>>(
      Hbuf, W2Ts, sh_b2, x, out, nullptr, nullptr, nullptr, 0, I, D);

  // experts: only routed tokens (top-2 gather), weighted scatter-add
  for (int e = 0; e < E; ++e) {
    moe_ffn1_k<true><<<dim3(Ntok / BM, I / BN), 256, 0, stream>>>(
        Xb, EW1T + (size_t)e * DI, eb1 + (size_t)e * I, Hbuf,
        tlist + (size_t)e * Ntok, counts + e, D, I);
    moe_ffn2_k<true><<<dim3(Ntok / BM, D / BN), 256, 0, stream>>>(
        Hbuf, EW2T + (size_t)e * DI, eb2 + (size_t)e * D, nullptr, out,
        tlist + (size_t)e * Ntok, counts + e, weights, e, I, D);
  }
}
